// MFPA_Module_23854248362752
// MI455X (gfx1250) — matrix operand order repaired and checked by reading, not compiled, not run on hardware
//
#include <hip/hip_runtime.h>
#include <hip/hip_bf16.h>

// ---------------------------------------------------------------------------
// MFPA module on gfx1250 (B=4, C=64, H=W=64 -> N=4096, CQK=16).
//
// Kernel 0: convert conv weights f32 -> f16 (tiny, one-shot).
// Kernel 1: WMMA projection GEMMs: one wave per 16-row tile of X^T;
//           12x v_wmma_f32_16x16x32_f16 per tile (Q:2, K:2, V:8).
//           -> Q [B][N][32] f16 (ch padded 16->32 with zeros)
//           -> K [B][N][32] f16 (same layout)
//           -> V [B][64][N] f16 (channel-major, so attn B-operands are b128s)
// Kernel 2: flash attention, one wave per (b, 16-query tile).
//           Per 32-key chunk: 2 WMMA (Q*K^T) + 4 WMMA (P*V^T).
//           Softmax: energies transposed through LDS; the A-operand key
//           pattern splits each row exactly between the two half-waves
//           (hi=0: keys {0-7,16-23}, hi=1: {8-15,24-31}), so each lane exps
//           only its 16 pattern positions and the row max/sum are merged
//           across halves with one __shfl_xor(16) each. P built in registers.
// ---------------------------------------------------------------------------

typedef __attribute__((ext_vector_type(16))) _Float16 v16h;
typedef __attribute__((ext_vector_type(8)))  float    v8f;

#define BB  4
#define CC  64
#define NN  4096
#define CQK 16

__device__ __forceinline__ int a_kpat(int t, int hi) {
    // 16-bit A-matrix 16x32 layout: VGPR t<4 holds K=2t(+8*hi), t>=4 holds
    // K=16+2(t-4)(+8*hi). Returns first K of the half-pair held in VGPR t.
    return (t < 4 ? 2 * t : 16 + 2 * (t - 4)) + hi * 8;
}

// ---------------------------------------------------------------------------
// Kernel 0: weight f32 -> f16 conversion (row-major [o][c], B-operand ready).
// ---------------------------------------------------------------------------
__global__ __launch_bounds__(256) void mfpa_cvtw_kernel(
    const float* __restrict__ Wq, const float* __restrict__ Wk,
    const float* __restrict__ Wv,
    _Float16* __restrict__ Wqh, _Float16* __restrict__ Wkh,
    _Float16* __restrict__ Wvh)
{
    int i = blockIdx.x * blockDim.x + threadIdx.x;
    if (i < CQK * CC) {
        Wqh[i] = (_Float16)Wq[i];
        Wkh[i] = (_Float16)Wk[i];
    }
    if (i < CC * CC) Wvh[i] = (_Float16)Wv[i];
}

// ---------------------------------------------------------------------------
// Kernel 1: WMMA projections. One wave per 16 spatial positions.
// ---------------------------------------------------------------------------
__global__ __launch_bounds__(32) void mfpa_proj_kernel(
    const float* __restrict__ x, const float* __restrict__ xr,
    const _Float16* __restrict__ Wqh, const float* __restrict__ bq,
    const _Float16* __restrict__ Wkh, const float* __restrict__ bk,
    const _Float16* __restrict__ Wvh, const float* __restrict__ bv,
    _Float16* __restrict__ Qh, _Float16* __restrict__ Kh,
    _Float16* __restrict__ Vh)
{
    const int tile = blockIdx.x;            // 0 .. B*(N/16)-1
    const int b    = tile >> 8;
    const int n0   = (tile & 255) << 4;
    const int m    = threadIdx.x & 15;
    const int hi   = threadIdx.x >> 4;

    // ---- A operands: X^T tile rows (n = n0+m), channels per A k-pattern.
    // x is [C][N]: fixed channel, lanes sweep contiguous n -> coalesced.
    v16h af0, af1, ar0, ar1;
    {
        const float* xb  = x  + (size_t)b * CC * NN + n0 + m;
        const float* xrb = xr + (size_t)b * CC * NN + n0 + m;
        #pragma unroll
        for (int t = 0; t < 8; ++t) {
            const int kb = a_kpat(t, hi);
            af0[2 * t]     = (_Float16)xb[(size_t)kb * NN];
            af0[2 * t + 1] = (_Float16)xb[(size_t)(kb + 1) * NN];
            af1[2 * t]     = (_Float16)xb[(size_t)(kb + 32) * NN];
            af1[2 * t + 1] = (_Float16)xb[(size_t)(kb + 33) * NN];
            ar0[2 * t]     = (_Float16)xrb[(size_t)kb * NN];
            ar0[2 * t + 1] = (_Float16)xrb[(size_t)(kb + 1) * NN];
            ar1[2 * t]     = (_Float16)xrb[(size_t)(kb + 32) * NN];
            ar1[2 * t + 1] = (_Float16)xrb[(size_t)(kb + 33) * NN];
        }
    }

    // B operand from weight rows: n = output (lane), k = channel chunk.
    auto loadB = [&](const _Float16* W, int ob, int cchunk) -> v16h {
        v16h vb;
        const _Float16* wr = W + (size_t)(ob * 16 + m) * CC + cchunk * 32 + hi * 8;
        #pragma unroll
        for (int t = 0; t < 8; ++t) {
            const int ko = t < 4 ? 2 * t : 2 * t + 8;
            vb[2 * t]     = wr[ko];
            vb[2 * t + 1] = wr[ko + 1];
        }
        return vb;
    };

    const v8f cz = {};

    // ---- K projection (16 outputs) ----
    {
        v8f d = __builtin_amdgcn_wmma_f32_16x16x32_f16(
                    false, af0, false, loadB(Wkh, 0, 0), (short)0, cz, false, false);
        d = __builtin_amdgcn_wmma_f32_16x16x32_f16(
                    false, af1, false, loadB(Wkh, 0, 1), (short)0, d, false, false);
        const float bv_ = bk[m];
        #pragma unroll
        for (int r = 0; r < 8; ++r) {
            _Float16* kr = Kh + ((size_t)b * NN + n0 + r + 8 * hi) * 32;
            kr[m]      = (_Float16)(d[r] + bv_);
            kr[16 + m] = (_Float16)0.0f;      // zero-pad channels 16..31
        }
    }
    // ---- Q projection (16 outputs, from x_RGB) ----
    {
        v8f d = __builtin_amdgcn_wmma_f32_16x16x32_f16(
                    false, ar0, false, loadB(Wqh, 0, 0), (short)0, cz, false, false);
        d = __builtin_amdgcn_wmma_f32_16x16x32_f16(
                    false, ar1, false, loadB(Wqh, 0, 1), (short)0, d, false, false);
        const float bv_ = bq[m];
        #pragma unroll
        for (int r = 0; r < 8; ++r) {
            _Float16* qr = Qh + ((size_t)b * NN + n0 + r + 8 * hi) * 32;
            qr[m]      = (_Float16)(d[r] + bv_);
            qr[16 + m] = (_Float16)0.0f;
        }
    }
    // ---- V projection (64 outputs = 4 blocks), stored channel-major ----
    #pragma unroll
    for (int ob = 0; ob < 4; ++ob) {
        v8f d = __builtin_amdgcn_wmma_f32_16x16x32_f16(
                    false, af0, false, loadB(Wvh, ob, 0), (short)0, cz, false, false);
        d = __builtin_amdgcn_wmma_f32_16x16x32_f16(
                    false, af1, false, loadB(Wvh, ob, 1), (short)0, d, false, false);
        const float bv_ = bv[ob * 16 + m];
        #pragma unroll
        for (int r = 0; r < 8; ++r) {
            Vh[((size_t)b * CC + ob * 16 + m) * NN + n0 + r + 8 * hi] =
                (_Float16)(d[r] + bv_);
        }
    }
}

// ---------------------------------------------------------------------------
// Kernel 2: flash attention, one wave per (b, 16-query tile).
// ---------------------------------------------------------------------------
#define EROW 36  // energy row stride in floats: 144 B -> 16B-aligned b128 reads

__global__ __launch_bounds__(32) void mfpa_attn_kernel(
    const _Float16* __restrict__ Qh, const _Float16* __restrict__ Kh,
    const _Float16* __restrict__ Vh, const float* __restrict__ x,
    const float* __restrict__ lam, float* __restrict__ out)
{
    __shared__ float Elds[16 * EROW];
    __shared__ float Slds[16];

    const int tile = blockIdx.x;
    const int b    = tile >> 8;
    const int i0   = (tile & 255) << 4;
    const int m    = threadIdx.x & 15;
    const int hi   = threadIdx.x >> 4;

    // ---- Q tile as WMMA A operand (resident all loop) ----
    v16h qa;
    {
        const _Float16* qrow = Qh + ((size_t)b * NN + i0 + m) * 32;
        #pragma unroll
        for (int t = 0; t < 8; ++t) {
            const int kb = a_kpat(t, hi);
            qa[2 * t]     = qrow[kb];
            qa[2 * t + 1] = qrow[kb + 1];
        }
    }

    float rmax = -3.0e38f, rsum = 0.0f;   // stats for row m (halves agree)
    v8f acc0 = {}, acc1 = {}, acc2 = {}, acc3 = {};

    const _Float16* kbase = Kh + (size_t)b * NN * 32;
    const _Float16* vbase = Vh + (size_t)b * CC * NN;

    for (int j0 = 0; j0 < NN; j0 += 32) {
        // ---- energy = Q*K^T for 32 keys (two 16-key WMMAs) ----
        v16h kb0, kb1;
        {
            const _Float16* kr0 = kbase + (size_t)(j0 + m) * 32 + hi * 8;
            const _Float16* kr1 = kbase + (size_t)(j0 + 16 + m) * 32 + hi * 8;
            #pragma unroll
            for (int t = 0; t < 8; ++t) {
                const int ko = t < 4 ? 2 * t : 2 * t + 8;
                kb0[2 * t]     = kr0[ko];
                kb0[2 * t + 1] = kr0[ko + 1];
                kb1[2 * t]     = kr1[ko];
                kb1[2 * t + 1] = kr1[ko + 1];
            }
        }
        const v8f cz = {};
        v8f e0 = __builtin_amdgcn_wmma_f32_16x16x32_f16(
                     false, qa, false, kb0, (short)0, cz, false, false);
        v8f e1 = __builtin_amdgcn_wmma_f32_16x16x32_f16(
                     false, qa, false, kb1, (short)0, cz, false, false);

        // ---- transpose energies through LDS (C-layout -> row-per-lane) ----
        #pragma unroll
        for (int r = 0; r < 8; ++r) {
            Elds[(r + 8 * hi) * EROW + m]      = e0[r];
            Elds[(r + 8 * hi) * EROW + 16 + m] = e1[r];
        }
        __syncthreads();

        // ---- softmax: lane handles the 16 key positions of row m that its
        // A-operand pattern needs ({0-7,16-23} for hi=0, {8-15,24-31} hi=1);
        // row max/sum are merged across halves with one shuffle each. ----
        float ev[16];
        const float* erow = &Elds[m * EROW];
        #pragma unroll
        for (int j = 0; j < 8; ++j) {
            ev[j]     = erow[hi * 8 + j];
            ev[8 + j] = erow[16 + hi * 8 + j];
        }
        float mxl = ev[0];
        #pragma unroll
        for (int j = 1; j < 16; ++j) mxl = fmaxf(mxl, ev[j]);
        const float mx = fmaxf(mxl, __shfl_xor(mxl, 16, 32));
        const float nm = fmaxf(rmax, mx);
        const float sc = __expf(rmax - nm);
        rmax = nm;
        float psl = 0.0f;
        #pragma unroll
        for (int j = 0; j < 16; ++j) { ev[j] = __expf(ev[j] - nm); psl += ev[j]; }
        const float ps = psl + __shfl_xor(psl, 16, 32);
        rsum = rsum * sc + ps;

        // ---- rescale accumulators only when the running max moved
        // (wave-uniform branch: EXEC stays all-ones). ----
        if (!__all(sc == 1.0f)) {
            Slds[m] = sc;                   // halves write identical values
            __syncthreads();
            float sc8[8];
            #pragma unroll
            for (int r = 0; r < 8; ++r) sc8[r] = Slds[hi * 8 + r];
            #pragma unroll
            for (int r = 0; r < 8; ++r) {
                acc0[r] *= sc8[r]; acc1[r] *= sc8[r];
                acc2[r] *= sc8[r]; acc3[r] *= sc8[r];
            }
        }

        // ---- P tile built directly in registers (A layout) ----
        v16h pa;
        #pragma unroll
        for (int t = 0; t < 4; ++t) {
            pa[2 * t]     = (_Float16)ev[2 * t];
            pa[2 * t + 1] = (_Float16)ev[2 * t + 1];
            pa[8 + 2 * t]     = (_Float16)ev[8 + 2 * t];
            pa[8 + 2 * t + 1] = (_Float16)ev[8 + 2 * t + 1];
        }

        // ---- O += P * V^T, 4 channel blocks ----
        #pragma unroll
        for (int cb = 0; cb < 4; ++cb) {
            v16h vb;
            const _Float16* vr = vbase + (size_t)(cb * 16 + m) * NN + j0 + hi * 8;
            #pragma unroll
            for (int t = 0; t < 8; ++t) {
                const int ko = t < 4 ? 2 * t : 2 * t + 8;
                vb[2 * t]     = vr[ko];
                vb[2 * t + 1] = vr[ko + 1];
            }
            if (cb == 0)
                acc0 = __builtin_amdgcn_wmma_f32_16x16x32_f16(
                           false, pa, false, vb, (short)0, acc0, false, false);
            else if (cb == 1)
                acc1 = __builtin_amdgcn_wmma_f32_16x16x32_f16(
                           false, pa, false, vb, (short)0, acc1, false, false);
            else if (cb == 2)
                acc2 = __builtin_amdgcn_wmma_f32_16x16x32_f16(
                           false, pa, false, vb, (short)0, acc2, false, false);
            else
                acc3 = __builtin_amdgcn_wmma_f32_16x16x32_f16(
                           false, pa, false, vb, (short)0, acc3, false, false);
        }
        __syncthreads();   // Elds/Slds reused next chunk
    }

    // ---- broadcast row sums to C-layout lanes, write output ----
    Slds[m] = rsum;
    __syncthreads();
    float rs8[8];
    #pragma unroll
    for (int r = 0; r < 8; ++r) rs8[r] = Slds[hi * 8 + r];

    const float lamv = lam[0];
    #pragma unroll
    for (int cb = 0; cb < 4; ++cb) {
        #pragma unroll
        for (int r = 0; r < 8; ++r) {
            const int i = i0 + r + 8 * hi;       // query index (M)
            const int c = cb * 16 + m;           // channel (N of block cb)
            const size_t off = ((size_t)b * CC + c) * NN + i;
            const float o = (cb == 0 ? acc0[r]
                           : cb == 1 ? acc1[r]
                           : cb == 2 ? acc2[r] : acc3[r]);
            out[off] = lamv * (o / rs8[r]) + x[off];
        }
    }
}

// ---------------------------------------------------------------------------
extern "C" void kernel_launch(void* const* d_in, const int* in_sizes, int n_in,
                              void* d_out, int out_size, void* d_ws, size_t ws_size,
                              hipStream_t stream) {
    const float* x   = (const float*)d_in[0];
    const float* xr  = (const float*)d_in[1];
    const float* Wq  = (const float*)d_in[2];
    const float* bq  = (const float*)d_in[3];
    const float* Wk  = (const float*)d_in[4];
    const float* bk  = (const float*)d_in[5];
    const float* Wv  = (const float*)d_in[6];
    const float* bv  = (const float*)d_in[7];
    const float* lam = (const float*)d_in[8];
    float* out = (float*)d_out;

    // workspace carve-up (f16): Q | K | V | Wq | Wk | Wv  (~4.01 MB)
    _Float16* Qh  = (_Float16*)d_ws;
    _Float16* Kh  = Qh  + (size_t)BB * NN * 32;
    _Float16* Vh  = Kh  + (size_t)BB * NN * 32;
    _Float16* Wqh = Vh  + (size_t)BB * CC * NN;
    _Float16* Wkh = Wqh + (size_t)CQK * CC;
    _Float16* Wvh = Wkh + (size_t)CQK * CC;

    mfpa_cvtw_kernel<<<(CC * CC + 255) / 256, 256, 0, stream>>>(
        Wq, Wk, Wv, Wqh, Wkh, Wvh);

    mfpa_proj_kernel<<<BB * (NN / 16), 32, 0, stream>>>(
        x, xr, Wqh, bq, Wkh, bk, Wvh, bv, Qh, Kh, Vh);

    mfpa_attn_kernel<<<BB * (NN / 16), 32, 0, stream>>>(
        Qh, Kh, Vh, x, lam, out);
}
